// HMMTagger_86904368267612
// MI455X (gfx1250) — matrix operand order repaired and checked by reading, not compiled, not run on hardware
//
#include <hip/hip_runtime.h>

// HMM forward-backward, rescaled to linear space so each scan step is a
// [16x128]@[128x128] GEMM done with v_wmma_f32_16x16x32_f16.
// B=64 sequences -> 4 workgroups of 16 rows (M=16 WMMA tile), 8 waves each
// owning one 16x16 N-tile with K=128 chained inside the wave.
// P tiles live in VGPRs for the whole 2048-step scan.
// Row reductions: VALU-only DPP16 xor-butterfly. A-fragment LDS loads are
// fenced with sched_barrier so they issue as one pipelined clause.

#define B_  64
#define S_  2048
#define T_  128
#define V_  50000

typedef __attribute__((ext_vector_type(16))) _Float16 v16h;
typedef __attribute__((ext_vector_type(8)))  _Float16 v8h;
typedef __attribute__((ext_vector_type(8)))  float    v8f;

// DPP16 xor-butterfly step: x += x_shuffled(ctrl). Masks {1,2,7,15} span the
// 16-lane group, so 4 steps give every lane the sum over its 16-lane row.
template <int CTRL>
__device__ __forceinline__ float dpp_add(float x) {
  int y = __builtin_amdgcn_update_dpp(0, __float_as_int(x), CTRL, 0xF, 0xF, true);
  return x + __int_as_float(y);
}
__device__ __forceinline__ float row16_sum(float x) {
  x = dpp_add<0xB1>(x);    // quad_perm [1,0,3,2]  : xor 1
  x = dpp_add<0x4E>(x);    // quad_perm [2,3,0,1]  : xor 2
  x = dpp_add<0x141>(x);   // row_half_mirror      : xor 7
  x = dpp_add<0x140>(x);   // row_mirror           : xor 15
  return x;
}

// Load the four 16x32 A fragments for this lane from row-major f16 LDS.
// Kept in distinct registers; sched_barrier keeps all 8 ds_load_b128 together.
__device__ __forceinline__ void load_a_frags(const _Float16* lds, int nl, int hi,
                                             v16h afr[4]) {
#pragma unroll
  for (int ch = 0; ch < 4; ++ch) {
    v8h alo = *(const v8h*)(lds + nl * T_ + 32 * ch + 16 * hi);
    v8h aup = *(const v8h*)(lds + nl * T_ + 32 * ch + 16 * hi + 8);
#pragma unroll
    for (int q = 0; q < 8; ++q) { afr[ch][q] = alo[q]; afr[ch][8 + q] = aup[q]; }
  }
  __builtin_amdgcn_sched_barrier(0);  // don't sink loads into the WMMA chain
}

// ---------------- prep: Pexp = exp(logT) (f16) and its transpose ----------------
__global__ void hmm_prep_P(const float* __restrict__ logT,
                           _Float16* __restrict__ Pexp,
                           _Float16* __restrict__ PexpT) {
  int idx = blockIdx.x * blockDim.x + threadIdx.x;  // 0..16383
  int i = idx >> 7, j = idx & 127;
  float v = __expf(logT[idx]);
  Pexp[idx]         = (_Float16)v;  // row-major P
  PexpT[j * T_ + i] = (_Float16)v;  // row-major P^T
}

// ------------- gather emissions: E[s][t][b] = exp(logE[t, wid[b,s]]) -------------
// Layout [S][T][B] so a WMMA C-fragment lane reads 8 consecutive floats (in b).
__global__ void hmm_gather_E(const float* __restrict__ logE,
                             const int* __restrict__ wid,
                             float* __restrict__ E) {
  __shared__ int w[16];
  int s = blockIdx.x, g = blockIdx.y;
  int tid = threadIdx.x;
  if (tid < 16) w[tid] = wid[(size_t)(16 * g + tid) * S_ + s];
  __syncthreads();
  int b = tid & 15, tq = tid >> 4;
#pragma unroll
  for (int p = 0; p < 8; ++p) {
    int t = p * 16 + tq;
    E[((size_t)s * T_ + t) * B_ + 16 * g + b] =
        __expf(logE[(size_t)t * V_ + w[b]]);
  }
}

// ---------------------------------- forward ----------------------------------
// alpha_s = normalize_rows( (alpha_{s-1} @ P) .* E_s ), stored f32 in [S][T][B].
__launch_bounds__(256, 1)
__global__ void hmm_forward(const float* __restrict__ log_prior,
                            const _Float16* __restrict__ PexpT,
                            const float* __restrict__ E,
                            float* __restrict__ alphaW) {
  __shared__ __align__(16) _Float16 aLds[16 * T_];   // alpha in A-matrix row-major f16
  __shared__ float partial[8 * 16];
  __shared__ __align__(16) float tot[16];

  const int g    = blockIdx.x;
  const int tid  = threadIdx.x;
  const int lane = tid & 31;
  const int wave = tid >> 5;
  const int nl   = lane & 15;
  const int hi   = lane >> 4;
  const int t    = wave * 16 + nl;       // this lane's C-fragment column (tag)
  const int bo   = 16 * g + 8 * hi;      // first of this lane's 8 batch rows

  // Resident B tiles: B[k][n]=P[k][t] -> contiguous row t of P^T. 32B/lane/chunk.
  v16h Bk[4];
#pragma unroll
  for (int c = 0; c < 4; ++c)
    Bk[c] = *(const v16h*)(PexpT + (size_t)t * T_ + 32 * c + 16 * hi);

  // ---- alpha_0 = prior .* E_0, row-normalized ----
  if (tid < 16) tot[tid] = 0.f;
  __syncthreads();
  for (int idx = tid; idx < 16 * T_; idx += 256) {
    int m = idx >> 7, tt = idx & 127;
    float v = __expf(log_prior[tt]) * E[(size_t)tt * B_ + 16 * g + m];
    atomicAdd(&tot[m], v);
  }
  __syncthreads();
  for (int idx = tid; idx < 16 * T_; idx += 256) {
    int m = idx >> 7, tt = idx & 127;
    float v = __expf(log_prior[tt]) * E[(size_t)tt * B_ + 16 * g + m];
    v /= tot[m];
    aLds[idx] = (_Float16)v;
    alphaW[(size_t)tt * B_ + 16 * g + m] = v;
  }
  __syncthreads();

  // double-buffered E prefetch (contiguous in b for this lane)
  const float4* Ep = (const float4*)(E + ((size_t)1 * T_ + t) * B_ + bo);
  float4 e0n = Ep[0], e1n = Ep[1];

  for (int s = 1; s < S_; ++s) {
    float4 e0 = e0n, e1 = e1n;
    {
      int sn = (s + 1 < S_) ? s + 1 : s;   // branchless prefetch
      const float4* En = (const float4*)(E + ((size_t)sn * T_ + t) * B_ + bo);
      e0n = En[0]; e1n = En[1];
    }

    // A fragments (one pipelined ds clause), then chained WMMAs.
    v16h afr[4];
    load_a_frags(aLds, nl, hi, afr);
    v8f c = {};
#pragma unroll
    for (int ch = 0; ch < 4; ++ch)
      c = __builtin_amdgcn_wmma_f32_16x16x32_f16(false, afr[ch], false, Bk[ch],
                                                 (short)0, c, false, false);

    // .* E_s, then per-row sums via DPP butterfly (VALU only)
    float v[8];
    v[0] = c[0] * e0.x; v[1] = c[1] * e0.y; v[2] = c[2] * e0.z; v[3] = c[3] * e0.w;
    v[4] = c[4] * e1.x; v[5] = c[5] * e1.y; v[6] = c[6] * e1.z; v[7] = c[7] * e1.w;

    float s8[8];
#pragma unroll
    for (int r = 0; r < 8; ++r) s8[r] = row16_sum(v[r]);
    if (nl == 0) {
#pragma unroll
      for (int r = 0; r < 8; ++r) partial[wave * 16 + 8 * hi + r] = s8[r];
    }
    __syncthreads();
    if (tid < 16) {
      float acc = 0.f;
#pragma unroll
      for (int w2 = 0; w2 < 8; ++w2) acc += partial[w2 * 16 + tid];
      tot[tid] = 1.0f / acc;
    }
    __syncthreads();
    float4 r0 = ((const float4*)(tot + 8 * hi))[0];
    float4 r1 = ((const float4*)(tot + 8 * hi))[1];
    float rc[8] = {r0.x, r0.y, r0.z, r0.w, r1.x, r1.y, r1.z, r1.w};

    float nv[8];
#pragma unroll
    for (int r = 0; r < 8; ++r) nv[r] = v[r] * rc[r];

    // normalized alpha -> workspace ([S][T][B], 2x16B per lane)
    float4* Aw = (float4*)(alphaW + ((size_t)s * T_ + t) * B_ + bo);
    Aw[0] = make_float4(nv[0], nv[1], nv[2], nv[3]);
    Aw[1] = make_float4(nv[4], nv[5], nv[6], nv[7]);

    // f16 copy to LDS for next step's A fragment
#pragma unroll
    for (int r = 0; r < 8; ++r)
      aLds[(8 * hi + r) * T_ + t] = (_Float16)nv[r];
    __syncthreads();
  }
}

// ---------------------------------- backward ----------------------------------
// beta_i = normalize_rows( (beta_{i+1} .* E_i) @ P^T ); out_i = normalize(alpha_i .* beta_i)
__launch_bounds__(256, 1)
__global__ void hmm_backward(const _Float16* __restrict__ Pexp,
                             const float* __restrict__ E,
                             const float* __restrict__ alphaW,
                             float* __restrict__ out) {
  __shared__ __align__(16) _Float16 vLds[16 * T_];
  __shared__ float partialB[8 * 16];
  __shared__ float partialP[8 * 16];
  __shared__ __align__(16) float totB[16];
  __shared__ __align__(16) float totP[16];

  const int g    = blockIdx.x;
  const int tid  = threadIdx.x;
  const int lane = tid & 31;
  const int wave = tid >> 5;
  const int nl   = lane & 15;
  const int hi   = lane >> 4;
  const int t    = wave * 16 + nl;
  const int bo   = 16 * g + 8 * hi;

  // Resident B tiles for P^T: B[k][n] = P[t][k] -> contiguous row t of P.
  v16h Bk[4];
#pragma unroll
  for (int c = 0; c < 4; ++c)
    Bk[c] = *(const v16h*)(Pexp + (size_t)t * T_ + 32 * c + 16 * hi);

  // marginal at S-1: beta=const cancels; alpha rows already sum to 1
  for (int idx = tid; idx < 16 * T_; idx += 256) {
    int m = idx >> 7, tt = idx & 127;
    out[((size_t)(16 * g + m) * S_ + (S_ - 1)) * T_ + tt] =
        alphaW[((size_t)(S_ - 1) * T_ + tt) * B_ + 16 * g + m];
  }

  float breg[8];
#pragma unroll
  for (int r = 0; r < 8; ++r) breg[r] = 1.0f;

  // prefetch E_i and alpha_i for i = S-2
  const float4* Ep = (const float4*)(E + ((size_t)(S_ - 2) * T_ + t) * B_ + bo);
  const float4* Ap = (const float4*)(alphaW + ((size_t)(S_ - 2) * T_ + t) * B_ + bo);
  float4 e0n = Ep[0], e1n = Ep[1], a0n = Ap[0], a1n = Ap[1];

  for (int i = S_ - 2; i >= 0; --i) {
    float4 e0 = e0n, e1 = e1n, a0 = a0n, a1 = a1n;
    {
      int in = (i > 0) ? i - 1 : 0;        // branchless prefetch
      const float4* En = (const float4*)(E + ((size_t)in * T_ + t) * B_ + bo);
      const float4* An = (const float4*)(alphaW + ((size_t)in * T_ + t) * B_ + bo);
      e0n = En[0]; e1n = En[1]; a0n = An[0]; a1n = An[1];
    }

    // v = beta_{i+1} .* E_i  -> A-matrix row-major f16 in LDS
    float vv[8];
    vv[0] = breg[0] * e0.x; vv[1] = breg[1] * e0.y;
    vv[2] = breg[2] * e0.z; vv[3] = breg[3] * e0.w;
    vv[4] = breg[4] * e1.x; vv[5] = breg[5] * e1.y;
    vv[6] = breg[6] * e1.z; vv[7] = breg[7] * e1.w;
#pragma unroll
    for (int r = 0; r < 8; ++r)
      vLds[(8 * hi + r) * T_ + t] = (_Float16)vv[r];
    __syncthreads();

    // beta_i = v @ P^T : pipelined A loads, then chained WMMAs
    v16h afr[4];
    load_a_frags(vLds, nl, hi, afr);
    v8f c = {};
#pragma unroll
    for (int ch = 0; ch < 4; ++ch)
      c = __builtin_amdgcn_wmma_f32_16x16x32_f16(false, afr[ch], false, Bk[ch],
                                                 (short)0, c, false, false);

    // two row reductions: sum(beta) for rescale, sum(alpha.*beta) for marginal
    float pr[8];
    pr[0] = c[0] * a0.x; pr[1] = c[1] * a0.y; pr[2] = c[2] * a0.z; pr[3] = c[3] * a0.w;
    pr[4] = c[4] * a1.x; pr[5] = c[5] * a1.y; pr[6] = c[6] * a1.z; pr[7] = c[7] * a1.w;

    float sb[8], sp[8];
#pragma unroll
    for (int r = 0; r < 8; ++r) {
      sb[r] = row16_sum(c[r]);
      sp[r] = row16_sum(pr[r]);
    }
    if (nl == 0) {
#pragma unroll
      for (int r = 0; r < 8; ++r) {
        partialB[wave * 16 + 8 * hi + r] = sb[r];
        partialP[wave * 16 + 8 * hi + r] = sp[r];
      }
    }
    __syncthreads();
    if (tid < 16) {
      float ab = 0.f, ap = 0.f;
#pragma unroll
      for (int w2 = 0; w2 < 8; ++w2) {
        ab += partialB[w2 * 16 + tid];
        ap += partialP[w2 * 16 + tid];
      }
      totB[tid] = 1.0f / ab;
      totP[tid] = 1.0f / ap;
    }
    __syncthreads();
    float4 rb0 = ((const float4*)(totB + 8 * hi))[0];
    float4 rb1 = ((const float4*)(totB + 8 * hi))[1];
    float4 rp0 = ((const float4*)(totP + 8 * hi))[0];
    float4 rp1 = ((const float4*)(totP + 8 * hi))[1];
    float rcB[8] = {rb0.x, rb0.y, rb0.z, rb0.w, rb1.x, rb1.y, rb1.z, rb1.w};
    float rcP[8] = {rp0.x, rp0.y, rp0.z, rp0.w, rp1.x, rp1.y, rp1.z, rp1.w};

#pragma unroll
    for (int r = 0; r < 8; ++r) {
      breg[r] = c[r] * rcB[r];                                      // next beta
      out[((size_t)(bo + r) * S_ + i) * T_ + t] = pr[r] * rcP[r];   // marginal
    }
  }
}

extern "C" void kernel_launch(void* const* d_in, const int* in_sizes, int n_in,
                              void* d_out, int out_size, void* d_ws, size_t ws_size,
                              hipStream_t stream) {
  const float* log_prior = (const float*)d_in[0];   // [T]
  const float* logT      = (const float*)d_in[1];   // [T,T]
  const float* logE      = (const float*)d_in[2];   // [T,V]
  const int*   wid       = (const int*)d_in[3];     // [B,S]
  float* out = (float*)d_out;                       // [B,S,T]

  char* ws = (char*)d_ws;
  _Float16* Pexp  = (_Float16*)ws;                               // 32 KB
  _Float16* PexpT = (_Float16*)(ws + 32 * 1024);                 // 32 KB
  float* E        = (float*)(ws + 64 * 1024);                    // 64 MB [S][T][B]
  float* alphaW   = (float*)(ws + 64 * 1024 +
                             (size_t)S_ * T_ * B_ * sizeof(float)); // 64 MB [S][T][B]

  hmm_prep_P<<<(T_ * T_) / 256, 256, 0, stream>>>(logT, Pexp, PexpT);
  dim3 gE(S_, B_ / 16);
  hmm_gather_E<<<gE, 256, 0, stream>>>(logE, wid, E);
  hmm_forward<<<B_ / 16, 256, 0, stream>>>(log_prior, PexpT, E, alphaW);
  hmm_backward<<<B_ / 16, 256, 0, stream>>>(Pexp, E, alphaW, out);
}
